// Se3EquivariantTransformer_14834817040537
// MI455X (gfx1250) — matrix operand order repaired and checked by reading, not compiled, not run on hardware
//
#include <hip/hip_runtime.h>
#include <math.h>

// SE(3)-equivariant transformer, MI455X (gfx1250, wave32, WMMA).
// Dead-code: out1/vec/Y1/Wv1 never reach the output -> dropped.
// Linearity: per-edge (h[src] @ W) hoisted to per-node GEMMs + gather.

typedef __bf16 bf16_t;
typedef __attribute__((ext_vector_type(16))) __bf16 v16bf;
typedef __attribute__((ext_vector_type(8)))  float  v8f;

#define WMMA_BF16(a, b, c) \
  __builtin_amdgcn_wmma_f32_16x16x32_bf16(false, (a), false, (b), (short)0, (c), false, false)

// A-matrix (16x32 bf16) per-lane K index: VGPR j holds K pair per ISA 7.12.2
__device__ __forceinline__ int kA(int j, int hi8) {
  return ((j >> 2) << 4) + hi8 + ((j & 3) << 1);
}
// B-matrix (32x16 bf16): same K order as A (lanes 0-15 K=0..7,16..23; lanes 16-31 K=8..15,24..31)
__device__ __forceinline__ int kB(int j, int hi) {
  return ((j >> 2) << 4) + (hi << 3) + ((j & 3) << 1);
}

// Sizes fixed by the reference
#define HCONST 4
#define FCONST 32
#define CCONST 64
#define DKCONST 32
#define CVCONST 16
#define RHCONST 16
#define BCONST 128
#define OUTCONST 64

// ---------------------------------------------------------------------------
// K0: init workspace (segment-max = -inf, accumulators = 0)
// ---------------------------------------------------------------------------
__global__ void init_kernel(float* __restrict__ m, int nm,
                            float* __restrict__ z, long long nz) {
  long long i = (long long)blockIdx.x * blockDim.x + threadIdx.x;
  if (i < nm) m[i] = -3.0e38f;
  if (i < nz) z[i] = 0.0f;
}

// ---------------------------------------------------------------------------
// K1: fused node GEMMs.  h = nf@W_emb+b_emb (K=32, 1 WMMA per tile), then
//     per-head Q/K/V0 = h@W (K=64, 2 accumulating WMMAs). 1 wave = 16 nodes.
// ---------------------------------------------------------------------------
__global__ __launch_bounds__(256) void node_proj_kernel(
    const float* __restrict__ nf,   const float* __restrict__ W_emb,
    const float* __restrict__ b_emb,const float* __restrict__ Wq,
    const float* __restrict__ Wk,   const float* __restrict__ Wv0,
    float* __restrict__ q_buf, float* __restrict__ k_buf,
    float* __restrict__ v0_buf, int N)
{
  __shared__ bf16_t h_lds[8][16][72];  // 72 stride: conflict-free 16-row reads

  const int wave = threadIdx.x >> 5;
  const int lane = threadIdx.x & 31;
  const int l15  = lane & 15;
  const int hi   = lane >> 4;
  const int hi8  = hi << 3;
  const int ntiles = (N + 15) >> 4;

  int tile = blockIdx.x * 8 + wave;
  const bool valid = (tile < ntiles);
  if (!valid) tile = 0;                 // keep EXEC all-ones for WMMA
  const int row0 = tile << 4;

  // ---- stage 1: h tile (16x64) = nf(16x32) @ W_emb(32x64) + b_emb ----
  v16bf a;
  {
    int r = row0 + l15; if (r >= N) r = N - 1;
    const float* rp = nf + (size_t)r * FCONST;
#pragma unroll
    for (int j = 0; j < 8; ++j) {
      int k = kA(j, hi8);
      a[2 * j]     = (bf16_t)rp[k];
      a[2 * j + 1] = (bf16_t)rp[k + 1];
    }
  }
#pragma unroll
  for (int t = 0; t < 4; ++t) {
    v16bf b;
#pragma unroll
    for (int j = 0; j < 8; ++j) {
      int k = kB(j, hi);
      b[2 * j]     = (bf16_t)W_emb[k * CCONST + t * 16 + l15];
      b[2 * j + 1] = (bf16_t)W_emb[(k + 1) * CCONST + t * 16 + l15];
    }
    v8f c = {};
    c = WMMA_BF16(a, b, c);
    float bias = b_emb[t * 16 + l15];
#pragma unroll
    for (int i = 0; i < 8; ++i)
      h_lds[wave][i + hi8][t * 16 + l15] = (bf16_t)(c[i] + bias);
  }
  __syncthreads();

  // ---- stage 2: A fragments of h (16x64 -> two 16x32 chunks) ----
  v16bf a0, a1;
#pragma unroll
  for (int j = 0; j < 8; ++j) {
    int k = kA(j, hi8);
    a0[2 * j]     = h_lds[wave][l15][k];
    a0[2 * j + 1] = h_lds[wave][l15][k + 1];
    a1[2 * j]     = h_lds[wave][l15][k + 32];
    a1[2 * j + 1] = h_lds[wave][l15][k + 33];
  }

#pragma unroll
  for (int hd = 0; hd < HCONST; ++hd) {
    const float* WqH = Wq  + (size_t)hd * CCONST * DKCONST;
    const float* WkH = Wk  + (size_t)hd * CCONST * DKCONST;
    const float* WvH = Wv0 + (size_t)hd * CCONST * CVCONST;

#pragma unroll
    for (int t = 0; t < 2; ++t) {   // DK = 32 -> two 16-col tiles
      // ---- q projection ----
      {
        v16bf b0, b1;
#pragma unroll
        for (int j = 0; j < 8; ++j) {
          int k = kB(j, hi);
          b0[2 * j]     = (bf16_t)WqH[k * DKCONST + t * 16 + l15];
          b0[2 * j + 1] = (bf16_t)WqH[(k + 1) * DKCONST + t * 16 + l15];
          b1[2 * j]     = (bf16_t)WqH[(k + 32) * DKCONST + t * 16 + l15];
          b1[2 * j + 1] = (bf16_t)WqH[(k + 33) * DKCONST + t * 16 + l15];
        }
        v8f c = {};
        c = WMMA_BF16(a0, b0, c);
        c = WMMA_BF16(a1, b1, c);
        if (valid) {
#pragma unroll
          for (int i = 0; i < 8; ++i) {
            int r = row0 + i + hi8;
            if (r < N)
              q_buf[(size_t)r * 128 + hd * 32 + t * 16 + l15] = c[i];
          }
        }
      }
      // ---- k projection ----
      {
        v16bf b0, b1;
#pragma unroll
        for (int j = 0; j < 8; ++j) {
          int k = kB(j, hi);
          b0[2 * j]     = (bf16_t)WkH[k * DKCONST + t * 16 + l15];
          b0[2 * j + 1] = (bf16_t)WkH[(k + 1) * DKCONST + t * 16 + l15];
          b1[2 * j]     = (bf16_t)WkH[(k + 32) * DKCONST + t * 16 + l15];
          b1[2 * j + 1] = (bf16_t)WkH[(k + 33) * DKCONST + t * 16 + l15];
        }
        v8f c = {};
        c = WMMA_BF16(a0, b0, c);
        c = WMMA_BF16(a1, b1, c);
        if (valid) {
#pragma unroll
          for (int i = 0; i < 8; ++i) {
            int r = row0 + i + hi8;
            if (r < N)
              k_buf[(size_t)r * 128 + hd * 32 + t * 16 + l15] = c[i];
          }
        }
      }
    }
    // ---- v0 projection (CV = 16 -> one tile) ----
    {
      v16bf b0, b1;
#pragma unroll
      for (int j = 0; j < 8; ++j) {
        int k = kB(j, hi);
        b0[2 * j]     = (bf16_t)WvH[k * CVCONST + l15];
        b0[2 * j + 1] = (bf16_t)WvH[(k + 1) * CVCONST + l15];
        b1[2 * j]     = (bf16_t)WvH[(k + 32) * CVCONST + l15];
        b1[2 * j + 1] = (bf16_t)WvH[(k + 33) * CVCONST + l15];
      }
      v8f c = {};
      c = WMMA_BF16(a0, b0, c);
      c = WMMA_BF16(a1, b1, c);
      if (valid) {
#pragma unroll
        for (int i = 0; i < 8; ++i) {
          int r = row0 + i + hi8;
          if (r < N)
            v0_buf[(size_t)r * 64 + hd * 16 + l15] = c[i];
        }
      }
    }
  }
}

// ---------------------------------------------------------------------------
// K2: per-edge scores + segment max.  score = <q[dst], k[src]*rk>/sqrt(DK)
// ---------------------------------------------------------------------------
__global__ __launch_bounds__(256) void edge_score_kernel(
    const float* __restrict__ pos, const int* __restrict__ ei,
    const float* __restrict__ R1,  const float* __restrict__ b1,
    const float* __restrict__ R2,  const float* __restrict__ b2,
    const float* __restrict__ q_buf, const float* __restrict__ k_buf,
    float* __restrict__ scores, float* __restrict__ mbuf, int E)
{
  __shared__ float R2s[HCONST * RHCONST * 64];  // 16 KB
  __shared__ float b2s[HCONST * 64];
  __shared__ float R1s[HCONST * RHCONST];
  __shared__ float b1s[HCONST * RHCONST];
  for (int i = threadIdx.x; i < HCONST * RHCONST * 64; i += 256) R2s[i] = R2[i];
  if (threadIdx.x < HCONST * 64) b2s[threadIdx.x] = b2[threadIdx.x];
  if (threadIdx.x < HCONST * RHCONST) {
    R1s[threadIdx.x] = R1[threadIdx.x];
    b1s[threadIdx.x] = b1[threadIdx.x];
  }
  __syncthreads();

  int e = blockIdx.x * 256 + threadIdx.x;
  if (e >= E) return;
  int src = ei[e];
  int dst = ei[E + e];

  __builtin_prefetch(q_buf + (size_t)dst * 128, 0, 0);  // global_prefetch_b8
  __builtin_prefetch(k_buf + (size_t)src * 128, 0, 0);

  float dx = pos[dst * 3 + 0] - pos[src * 3 + 0];
  float dy = pos[dst * 3 + 1] - pos[src * 3 + 1];
  float dz = pos[dst * 3 + 2] - pos[src * 3 + 2];
  float d = sqrtf(dx * dx + dy * dy + dz * dz) + 1e-8f;

  float sc[HCONST];
#pragma unroll
  for (int hd = 0; hd < HCONST; ++hd) {
    float rh[RHCONST];
#pragma unroll
    for (int r = 0; r < RHCONST; ++r) {
      float x = d * R1s[hd * RHCONST + r] + b1s[hd * RHCONST + r];
      rh[r] = x / (1.0f + __expf(-x));   // silu
    }
    const float* qd = q_buf + (size_t)dst * 128 + hd * 32;
    const float* ks = k_buf + (size_t)src * 128 + hd * 32;
    float s = 0.0f;
#pragma unroll 4
    for (int o = 0; o < DKCONST; ++o) {
      float rk = b2s[hd * 64 + o];
#pragma unroll
      for (int r = 0; r < RHCONST; ++r)
        rk = fmaf(rh[r], R2s[hd * 1024 + r * 64 + o], rk);
      s = fmaf(qd[o] * ks[o], rk, s);
    }
    sc[hd] = s * 0.17677669529663687f;   // 1/sqrt(32)
    __hip_atomic_fetch_max(&mbuf[(size_t)dst * HCONST + hd], sc[hd],
                           __ATOMIC_RELAXED, __HIP_MEMORY_SCOPE_AGENT);
  }
  *(float4*)(scores + (size_t)e * HCONST) = make_float4(sc[0], sc[1], sc[2], sc[3]);
}

// ---------------------------------------------------------------------------
// K3: ex = exp(score - m[dst]); denom[dst] += ex  (scores overwritten by ex)
// ---------------------------------------------------------------------------
__global__ __launch_bounds__(256) void edge_softmax_kernel(
    const int* __restrict__ ei, float* __restrict__ scores,
    const float* __restrict__ mbuf, float* __restrict__ denom, int E)
{
  int e = blockIdx.x * 256 + threadIdx.x;
  if (e >= E) return;
  int dst = ei[E + e];
  float4 s = *(const float4*)(scores + (size_t)e * HCONST);
  float4 m = *(const float4*)(mbuf + (size_t)dst * HCONST);
  float4 ex;
  ex.x = __expf(s.x - m.x);
  ex.y = __expf(s.y - m.y);
  ex.z = __expf(s.z - m.z);
  ex.w = __expf(s.w - m.w);
  *(float4*)(scores + (size_t)e * HCONST) = ex;
  atomicAdd(&denom[(size_t)dst * HCONST + 0], ex.x);
  atomicAdd(&denom[(size_t)dst * HCONST + 1], ex.y);
  atomicAdd(&denom[(size_t)dst * HCONST + 2], ex.z);
  atomicAdd(&denom[(size_t)dst * HCONST + 3], ex.w);
}

// ---------------------------------------------------------------------------
// K4: alpha = ex/(denom+1e-9); out0[dst] += alpha * v0[src] * rv0(d)
// ---------------------------------------------------------------------------
__global__ __launch_bounds__(256) void edge_accum_kernel(
    const float* __restrict__ pos, const int* __restrict__ ei,
    const float* __restrict__ R1,  const float* __restrict__ b1,
    const float* __restrict__ R2,  const float* __restrict__ b2,
    const float* __restrict__ v0_buf, const float* __restrict__ ex_buf,
    const float* __restrict__ denom, float* __restrict__ out0, int E)
{
  __shared__ float R2v[HCONST * RHCONST * CVCONST];  // rv0 cols only (32..47)
  __shared__ float b2v[HCONST * CVCONST];
  __shared__ float R1s[HCONST * RHCONST];
  __shared__ float b1s[HCONST * RHCONST];
  for (int i = threadIdx.x; i < HCONST * RHCONST * CVCONST; i += 256) {
    int hd = i >> 8, r = (i >> 4) & 15, v = i & 15;
    R2v[i] = R2[hd * 1024 + r * 64 + 32 + v];
  }
  if (threadIdx.x < HCONST * CVCONST) {
    int hd = threadIdx.x >> 4, v = threadIdx.x & 15;
    b2v[threadIdx.x] = b2[hd * 64 + 32 + v];
  }
  if (threadIdx.x < HCONST * RHCONST) {
    R1s[threadIdx.x] = R1[threadIdx.x];
    b1s[threadIdx.x] = b1[threadIdx.x];
  }
  __syncthreads();

  int e = blockIdx.x * 256 + threadIdx.x;
  if (e >= E) return;
  int src = ei[e];
  int dst = ei[E + e];

  float dx = pos[dst * 3 + 0] - pos[src * 3 + 0];
  float dy = pos[dst * 3 + 1] - pos[src * 3 + 1];
  float dz = pos[dst * 3 + 2] - pos[src * 3 + 2];
  float d = sqrtf(dx * dx + dy * dy + dz * dz) + 1e-8f;

  float4 ex4 = *(const float4*)(ex_buf + (size_t)e * HCONST);
  float4 dn4 = *(const float4*)(denom + (size_t)dst * HCONST);
  float alpha[HCONST] = {ex4.x / (dn4.x + 1e-9f), ex4.y / (dn4.y + 1e-9f),
                         ex4.z / (dn4.z + 1e-9f), ex4.w / (dn4.w + 1e-9f)};

#pragma unroll
  for (int hd = 0; hd < HCONST; ++hd) {
    float rh[RHCONST];
#pragma unroll
    for (int r = 0; r < RHCONST; ++r) {
      float x = d * R1s[hd * RHCONST + r] + b1s[hd * RHCONST + r];
      rh[r] = x / (1.0f + __expf(-x));
    }
    const float* vsrc = v0_buf + (size_t)src * 64 + hd * 16;
    float* odst = out0 + (size_t)dst * 64 + hd * 16;
#pragma unroll 4
    for (int v = 0; v < CVCONST; ++v) {
      float rv0 = b2v[hd * 16 + v];
#pragma unroll
      for (int r = 0; r < RHCONST; ++r)
        rv0 = fmaf(rh[r], R2v[hd * 256 + r * 16 + v], rv0);
      atomicAdd(&odst[v], alpha[hd] * vsrc[v] * rv0);
    }
  }
}

// ---------------------------------------------------------------------------
// K5: batch pooling (segment sum over sorted batch ids, via atomics)
// ---------------------------------------------------------------------------
__global__ __launch_bounds__(256) void pool_kernel(
    const int* __restrict__ batch, const float* __restrict__ out0,
    float* __restrict__ pooled, int N)
{
  long long idx = (long long)blockIdx.x * 256 + threadIdx.x;
  if (idx >= (long long)N * 64) return;
  int n = (int)(idx >> 6);
  int c = (int)(idx & 63);
  atomicAdd(&pooled[(size_t)batch[n] * 64 + c], out0[idx]);
}

// ---------------------------------------------------------------------------
// K6: out(128x64) = pooled(128x64) @ Wproj(64x64), WMMA bf16, one block
// ---------------------------------------------------------------------------
__global__ __launch_bounds__(256) void final_gemm_kernel(
    const float* __restrict__ pooled, const float* __restrict__ Wproj,
    float* __restrict__ out)
{
  const int wave = threadIdx.x >> 5;
  const int lane = threadIdx.x & 31;
  const int l15  = lane & 15;
  const int hi   = lane >> 4;
  const int hi8  = hi << 3;
  const int row0 = wave << 4;   // 8 waves x 16 rows = 128 = B

  v16bf a0, a1;
  const float* rp = pooled + (size_t)(row0 + l15) * 64;
#pragma unroll
  for (int j = 0; j < 8; ++j) {
    int k = kA(j, hi8);
    a0[2 * j]     = (bf16_t)rp[k];
    a0[2 * j + 1] = (bf16_t)rp[k + 1];
    a1[2 * j]     = (bf16_t)rp[k + 32];
    a1[2 * j + 1] = (bf16_t)rp[k + 33];
  }
#pragma unroll
  for (int t = 0; t < 4; ++t) {
    v16bf b0, b1;
#pragma unroll
    for (int j = 0; j < 8; ++j) {
      int k = kB(j, hi);
      b0[2 * j]     = (bf16_t)Wproj[k * OUTCONST + t * 16 + l15];
      b0[2 * j + 1] = (bf16_t)Wproj[(k + 1) * OUTCONST + t * 16 + l15];
      b1[2 * j]     = (bf16_t)Wproj[(k + 32) * OUTCONST + t * 16 + l15];
      b1[2 * j + 1] = (bf16_t)Wproj[(k + 33) * OUTCONST + t * 16 + l15];
    }
    v8f c = {};
    c = WMMA_BF16(a0, b0, c);
    c = WMMA_BF16(a1, b1, c);
#pragma unroll
    for (int i = 0; i < 8; ++i)
      out[(size_t)(row0 + i + hi8) * OUTCONST + t * 16 + l15] = c[i];
  }
}

// ---------------------------------------------------------------------------
extern "C" void kernel_launch(void* const* d_in, const int* in_sizes, int n_in,
                              void* d_out, int out_size, void* d_ws, size_t ws_size,
                              hipStream_t stream) {
  const float* pos   = (const float*)d_in[0];
  const float* nf    = (const float*)d_in[1];
  const int*   ei    = (const int*)d_in[2];
  const int*   batch = (const int*)d_in[3];
  const float* W_emb = (const float*)d_in[4];
  const float* b_emb = (const float*)d_in[5];
  const float* Wq    = (const float*)d_in[6];
  const float* Wk    = (const float*)d_in[7];
  const float* Wv0   = (const float*)d_in[8];
  // d_in[9] = Wv1: dead code (out1 channels never reach the output)
  const float* R1    = (const float*)d_in[10];
  const float* b1    = (const float*)d_in[11];
  const float* R2    = (const float*)d_in[12];
  const float* b2    = (const float*)d_in[13];
  const float* Wproj = (const float*)d_in[14];

  const int N = in_sizes[0] / 3;
  const int E = in_sizes[2] / 2;

  // workspace layout (floats): q | k | v0 | scores | m | [denom out0 pooled]
  float* ws = (float*)d_ws;
  size_t off = 0;
  float* q_buf  = ws + off; off += (size_t)N * 128;
  float* k_buf  = ws + off; off += (size_t)N * 128;
  float* v0_buf = ws + off; off += (size_t)N * 64;
  float* scores = ws + off; off += (size_t)E * HCONST;
  float* mbuf   = ws + off; off += (size_t)N * HCONST;
  float* denom  = ws + off; off += (size_t)N * HCONST;
  float* out0   = ws + off; off += (size_t)N * 64;
  float* pooled = ws + off; off += (size_t)BCONST * 64;

  const long long nz = (long long)N * HCONST + (long long)N * 64 + BCONST * 64;
  {
    long long mx = nz > (long long)N * HCONST ? nz : (long long)N * HCONST;
    int blocks = (int)((mx + 255) / 256);
    init_kernel<<<blocks, 256, 0, stream>>>(mbuf, N * HCONST, denom, nz);
  }

  {
    int ntiles = (N + 15) / 16;
    int blocks = (ntiles + 7) / 8;
    node_proj_kernel<<<blocks, 256, 0, stream>>>(nf, W_emb, b_emb, Wq, Wk, Wv0,
                                                 q_buf, k_buf, v0_buf, N);
  }

  int eblocks = (E + 255) / 256;
  edge_score_kernel<<<eblocks, 256, 0, stream>>>(pos, ei, R1, b1, R2, b2,
                                                 q_buf, k_buf, scores, mbuf, E);
  edge_softmax_kernel<<<eblocks, 256, 0, stream>>>(ei, scores, mbuf, denom, E);
  edge_accum_kernel<<<eblocks, 256, 0, stream>>>(pos, ei, R1, b1, R2, b2,
                                                 v0_buf, scores, denom, out0, E);

  {
    long long total = (long long)N * 64;
    int blocks = (int)((total + 255) / 256);
    pool_kernel<<<blocks, 256, 0, stream>>>(batch, out0, pooled, N);
  }

  final_gemm_kernel<<<1, 256, 0, stream>>>(pooled, Wproj, (float*)d_out);
}
